// Critic_att_66975720013854
// MI455X (gfx1250) — matrix operand order repaired and checked by reading, not compiled, not run on hardware
//
#include <hip/hip_runtime.h>

// ---------------------------------------------------------------------------
// Critic-with-attention for MI455X (gfx1250, wave32, WMMA).
// All GEMMs run as v_wmma_f32_16x16x32_f16 (f16 inputs, f32 accumulate).
// Kernel 1: actor MLP, one wave = 16 batch-agent rows, weights + tiles in LDS.
// Kernel 2: critic attention + MLP, one wave = 16 batch items.
// Round 3: bias values pre-gathered into registers (no per-call LDS bias load,
//          no null-check branches); activation folded via template.
// ---------------------------------------------------------------------------

#define AIDX      3
#define BATCH_N   16384
#define NAG       16
#define LOBS      64
#define INV_SCALE 0.25f
#define LN_EPS    1e-5f

typedef __attribute__((ext_vector_type(16))) _Float16 v16h;
typedef __attribute__((ext_vector_type(8)))  float    v8f;

__device__ __forceinline__ int lane_id() { return (int)(threadIdx.x & 31); }

// ---- WMMA fragment builders (branch-free gathers from LDS, f32 -> f16) ----
// A fragment (16xK, M rows): A[m][k] = src[m*ld + coloff + k]
template <int KMAX>
__device__ __forceinline__ v16h frag_A(const float* src, int ld, int coloff) {
  const int l = lane_id(), m = l & 15, half = l >> 4;
  v16h a;
#pragma unroll
  for (int e = 0; e < 16; ++e) {
    const int k  = e + half * 8 + ((e >= 8) ? 8 : 0);
    const int kk = (k < KMAX) ? k : 0;            // clamp: load is always in-bounds
    const float v = src[m * ld + coloff + kk];    // unconditional ds_load
    a[e] = (_Float16)((k < KMAX) ? v : 0.0f);     // v_cndmask, no branch
  }
  return a;
}
// A fragment, transposed source: A[m][k] = src[k*ld + moff + m]
template <int KMAX>
__device__ __forceinline__ v16h frag_AT(const float* src, int ld, int moff) {
  const int l = lane_id(), m = l & 15, half = l >> 4;
  v16h a;
#pragma unroll
  for (int e = 0; e < 16; ++e) {
    const int k  = e + half * 8 + ((e >= 8) ? 8 : 0);
    const int kk = (k < KMAX) ? k : 0;
    const float v = src[kk * ld + moff + m];
    a[e] = (_Float16)((k < KMAX) ? v : 0.0f);
  }
  return a;
}
// B fragment (KxN, N=16): B[k][n] = src[(k+koff)*ld + noff + n]
template <int KMAX>
__device__ __forceinline__ v16h frag_B(const float* src, int ld, int koff, int noff) {
  const int l = lane_id(), n = l & 15, half = l >> 4;
  v16h b;
#pragma unroll
  for (int e = 0; e < 16; ++e) {
    const int k  = e + half * 8 + ((e >= 8) ? 8 : 0);
    const int kk = (k < KMAX) ? k : 0;
    const float v = src[(kk + koff) * ld + noff + n];
    b[e] = (_Float16)((k < KMAX) ? v : 0.0f);
  }
  return b;
}
// B fragment, transposed source: B[k][n] = src[n*ld + koff + k]
template <int KMAX>
__device__ __forceinline__ v16h frag_BT(const float* src, int ld, int koff) {
  const int l = lane_id(), n = l & 15, half = l >> 4;
  v16h b;
#pragma unroll
  for (int e = 0; e < 16; ++e) {
    const int k  = e + half * 8 + ((e >= 8) ? 8 : 0);
    const int kk = (k < KMAX) ? k : 0;
    const float v = src[n * ld + koff + kk];
    b[e] = (_Float16)((k < KMAX) ? v : 0.0f);
  }
  return b;
}

__device__ __forceinline__ v8f wmma16(v16h a, v16h b, v8f c) {
  return __builtin_amdgcn_wmma_f32_16x16x32_f16(false, a, false, b, (short)0, c, false, false);
}

// Store C tile (16x16 f32) with a pre-gathered per-lane bias value.
// ACT: 0=none, 1=relu, 2=leaky_relu(0.01)
template <int ACT>
__device__ __forceinline__ void store_C(float* dst, int ld, int noff, v8f c, float bb) {
  const int l = lane_id(), n = l & 15, half = l >> 4;
#pragma unroll
  for (int v = 0; v < 8; ++v) {
    float x = c[v] + bb;
    if (ACT == 1)      x = fmaxf(x, 0.0f);
    else if (ACT == 2) x = (x > 0.0f) ? x : 0.01f * x;
    dst[(v + half * 8) * ld + noff + n] = x;
  }
}

// ---------------------------------------------------------------------------
// Actor params
// ---------------------------------------------------------------------------
struct ActorParams {
  const float *Wself1, *bself1, *Wself2, *bself2;
  const float *Wfood1, *bfood1, *Wfood2, *bfood2;
  const float *Woth1,  *both1,  *Woth2,  *both2;
  const float *Wm1, *bm1, *Wm2, *bm2, *Wmo, *bmo;
  const float *flng, *flnb, *olng, *olnb;
};

// Actor weight LDS offsets (floats)
#define AO_WSELF1 0
#define AO_BSELF1 128
#define AO_WSELF2 160
#define AO_BSELF2 672
#define AO_WFOOD1 688
#define AO_BFOOD1 752
#define AO_WFOOD2 784
#define AO_BFOOD2 1296
#define AO_WOTH1  1312
#define AO_BOTH1  1376
#define AO_WOTH2  1408
#define AO_BOTH2  1920
#define AO_WM1    1936
#define AO_BM1    3472
#define AO_WM2    3504
#define AO_BM2    4528
#define AO_WMO    4560
#define AO_BMO    5072
#define AO_FLNG   5088
#define AO_FLNB   5104
#define AO_OLNG   5120
#define AO_OLNB   5136
#define AO_WTOT   5184
#define AO_PWAVE  5952   // per-wave scratch floats

// softmax-attention + LN over a per-row feature set stored as f16 [16][16][16]
__device__ void wave_attention(const float* S, const _Float16* FF, int F,
                               float* SC, float* FA, float* FO,
                               float* OUT, int outLd, int outOff,
                               const float* g, const float* bvec) {
  const int l = lane_id(), row = l & 15, half = l >> 4;
  // scores
  for (int j = 0; j < 8; ++j) {
    int f = half * 8 + j;
    if (f < F) {
      float acc = 0.0f;
#pragma unroll
      for (int d = 0; d < 16; ++d)
        acc += S[row * 16 + d] * (float)FF[(row * 16 + f) * 16 + d];
      SC[row * 16 + f] = acc * INV_SCALE;
    }
  }
  __syncthreads();
  if (half == 0) {
    float mx = -1e30f;
    for (int f = 0; f < F; ++f) mx = fmaxf(mx, SC[row * 16 + f]);
    float s = 0.0f;
    for (int f = 0; f < F; ++f) { float e = __expf(SC[row * 16 + f] - mx); FA[row * 16 + f] = e; s += e; }
    float inv = 1.0f / s;
    for (int f = 0; f < F; ++f) FA[row * 16 + f] *= inv;
  }
  __syncthreads();
  for (int j = 0; j < 8; ++j) {
    int d = half * 8 + j;
    float acc = 0.0f;
    for (int f = 0; f < F; ++f)
      acc += FA[row * 16 + f] * (float)FF[(row * 16 + f) * 16 + d];
    FO[row * 16 + d] = acc;
  }
  __syncthreads();
  if (half == 0) {
    float m = 0.0f, v = 0.0f;
    for (int d = 0; d < 16; ++d) m += FO[row * 16 + d];
    m *= (1.0f / 16.0f);
    for (int d = 0; d < 16; ++d) { float t = FO[row * 16 + d] - m; v += t * t; }
    v *= (1.0f / 16.0f);
    float inv = rsqrtf(v + LN_EPS);
    for (int d = 0; d < 16; ++d) {
      float t = (FO[row * 16 + d] - m) * inv * g[d] + bvec[d];
      OUT[row * outLd + outOff + d] = fmaxf(t, 0.0f);
    }
  }
  __syncthreads();
}

extern __shared__ float smem[];

__global__ __launch_bounds__(128) void actor_kernel(const float* __restrict__ obs,
                                                    const float* __restrict__ act,
                                                    ActorParams P,
                                                    float* __restrict__ wsO) {
  float* W = smem;
  const int tid = threadIdx.x;
  // stage actor weights into LDS (cooperative)
  {
    const float* srcs[22] = {P.Wself1,P.bself1,P.Wself2,P.bself2,P.Wfood1,P.bfood1,
                             P.Wfood2,P.bfood2,P.Woth1,P.both1,P.Woth2,P.both2,
                             P.Wm1,P.bm1,P.Wm2,P.bm2,P.Wmo,P.bmo,
                             P.flng,P.flnb,P.olng,P.olnb};
    const int offs[22] = {AO_WSELF1,AO_BSELF1,AO_WSELF2,AO_BSELF2,AO_WFOOD1,AO_BFOOD1,
                          AO_WFOOD2,AO_BFOOD2,AO_WOTH1,AO_BOTH1,AO_WOTH2,AO_BOTH2,
                          AO_WM1,AO_BM1,AO_WM2,AO_BM2,AO_WMO,AO_BMO,
                          AO_FLNG,AO_FLNB,AO_OLNG,AO_OLNB};
    const int ns[22]   = {128,32,512,16,64,32,512,16,64,32,512,16,1536,32,1024,32,512,16,16,16,16,16};
    for (int s = 0; s < 22; ++s)
      for (int i = tid; i < ns[s]; i += blockDim.x) W[offs[s] + i] = srcs[s][i];
  }
  __syncthreads();

  const int wave = tid >> 5;
  float* PWB = smem + AO_WTOT + wave * AO_PWAVE;
  float* X   = PWB;                 // [16][68] xin tile
  float* S   = PWB + 1088;          // [16][16] self features
  float* T0  = PWB + 1344;          // [16][32]
  float* T1  = PWB + 1856;          // [16][32]
  float* M0  = PWB + 2368;          // [16][48] merged features
  float* SC  = PWB + 3136;          // [16][16] scores
  float* FA  = PWB + 3392;          // [16][16] softmax
  float* FO  = PWB + 3648;          // [16][16] weighted sum
  _Float16* FF = (_Float16*)(PWB + 3904);  // [16][16][16] f16 feature sets

  const int tile = blockIdx.x * 4 + wave;        // 16384 tiles of 16 rows
  const int m0   = tile * 16;
  const int agt  = m0 >> 14;                     // m0 / BATCH_N
  const int b0   = m0 & (BATCH_N - 1);
  const int l = lane_id(), lrow = l & 15, lhalf = l >> 4;

  // build xin tile: [obs[b, a*64 : a*64+64], act[b, 2a], act[b, 2a+1]]
  for (int i = l; i < 16 * 66; i += 32) {
    int r = i / 66, c = i - r * 66;
    float v;
    if (c < 64) v = obs[(size_t)(b0 + r) * (NAG * LOBS) + agt * LOBS + c];
    else        v = act[(size_t)(b0 + r) * (NAG * 2) + agt * 2 + (c - 64)];
    X[r * 68 + c] = v;
  }
  __syncthreads();
  if (lhalf == 0) {
    S[lrow * 16 + 0] = X[lrow * 68 + 0];
    S[lrow * 16 + 1] = X[lrow * 68 + 1];
    S[lrow * 16 + 2] = X[lrow * 68 + 64];
    S[lrow * 16 + 3] = X[lrow * 68 + 65];
  }
  __syncthreads();

  // --- self MLP: 4 -> 32 -> 16 ---
  {
    const float bs1_0 = W[AO_BSELF1 + lrow], bs1_1 = W[AO_BSELF1 + 16 + lrow];
    const float bs2   = W[AO_BSELF2 + lrow];
    v16h A = frag_A<4>(S, 16, 0);
    v8f c0 = {}, c1 = {};
    c0 = wmma16(A, frag_B<4>(W + AO_WSELF1, 32, 0, 0), c0);
    c1 = wmma16(A, frag_B<4>(W + AO_WSELF1, 32, 0, 16), c1);
    store_C<1>(T0, 32, 0,  c0, bs1_0);
    store_C<1>(T0, 32, 16, c1, bs1_1);
    __syncthreads();
    v16h A2 = frag_A<32>(T0, 32, 0);
    v8f c = {};
    c = wmma16(A2, frag_B<32>(W + AO_WSELF2, 16, 0, 0), c);
    store_C<1>(S, 16, 0, c, bs2);
    store_C<1>(M0, 48, 0, c, bs2);
    __syncthreads();
  }

  // --- food features: 16 foods, 2 -> 32 -> 16 each ---
  {
    // hoist loop-invariant weight fragments + biases into registers
    const v16h Bf1a = frag_B<2>(W + AO_WFOOD1, 32, 0, 0);
    const v16h Bf1b = frag_B<2>(W + AO_WFOOD1, 32, 0, 16);
    const v16h Bf2  = frag_B<32>(W + AO_WFOOD2, 16, 0, 0);
    const float b1_0 = W[AO_BFOOD1 + lrow], b1_1 = W[AO_BFOOD1 + 16 + lrow];
    const float b2   = W[AO_BFOOD2 + lrow];
    for (int ft = 0; ft < 16; ++ft) {
      v16h A = frag_A<2>(X, 68, 2 + 2 * ft);
      v8f c0 = {}, c1 = {};
      c0 = wmma16(A, Bf1a, c0);
      c1 = wmma16(A, Bf1b, c1);
      store_C<1>(T0, 32, 0,  c0, b1_0);
      store_C<1>(T0, 32, 16, c1, b1_1);
      __syncthreads();
      v16h A2 = frag_A<32>(T0, 32, 0);
      v8f c = {};
      c = wmma16(A2, Bf2, c);
#pragma unroll
      for (int v = 0; v < 8; ++v) {
        float x = fmaxf(c[v] + b2, 0.0f);
        FF[((v + lhalf * 8) * 16 + ft) * 16 + lrow] = (_Float16)x;
      }
      __syncthreads();
    }
  }
  wave_attention(S, FF, 16, SC, FA, FO, M0, 48, 16, W + AO_FLNG, W + AO_FLNB);

  // --- other-agent features: 15 others, 2 -> 32 -> 16 each ---
  {
    const v16h Bo1a = frag_B<2>(W + AO_WOTH1, 32, 0, 0);
    const v16h Bo1b = frag_B<2>(W + AO_WOTH1, 32, 0, 16);
    const v16h Bo2  = frag_B<32>(W + AO_WOTH2, 16, 0, 0);
    const float b1_0 = W[AO_BOTH1 + lrow], b1_1 = W[AO_BOTH1 + 16 + lrow];
    const float b2   = W[AO_BOTH2 + lrow];
    for (int ot = 0; ot < 15; ++ot) {
      v16h A = frag_A<2>(X, 68, 34 + 2 * ot);
      v8f c0 = {}, c1 = {};
      c0 = wmma16(A, Bo1a, c0);
      c1 = wmma16(A, Bo1b, c1);
      store_C<1>(T0, 32, 0,  c0, b1_0);
      store_C<1>(T0, 32, 16, c1, b1_1);
      __syncthreads();
      v16h A2 = frag_A<32>(T0, 32, 0);
      v8f c = {};
      c = wmma16(A2, Bo2, c);
#pragma unroll
      for (int v = 0; v < 8; ++v) {
        float x = fmaxf(c[v] + b2, 0.0f);
        FF[((v + lhalf * 8) * 16 + ot) * 16 + lrow] = (_Float16)x;
      }
      __syncthreads();
    }
  }
  wave_attention(S, FF, 15, SC, FA, FO, M0, 48, 32, W + AO_OLNG, W + AO_OLNB);

  // --- merged MLP: 48 -> 32 -> 32 -> 16 ---
  {
    const float bm1_0 = W[AO_BM1 + lrow], bm1_1 = W[AO_BM1 + 16 + lrow];
    const float bm2_0 = W[AO_BM2 + lrow], bm2_1 = W[AO_BM2 + 16 + lrow];
    const float bmo   = W[AO_BMO + lrow];
    v16h Alo = frag_A<32>(M0, 48, 0);
    v16h Ahi = frag_A<16>(M0, 48, 32);
    v8f c0 = {}, c1 = {};
    c0 = wmma16(Alo, frag_B<32>(W + AO_WM1, 32, 0,  0), c0);
    c0 = wmma16(Ahi, frag_B<16>(W + AO_WM1, 32, 32, 0), c0);
    c1 = wmma16(Alo, frag_B<32>(W + AO_WM1, 32, 0,  16), c1);
    c1 = wmma16(Ahi, frag_B<16>(W + AO_WM1, 32, 32, 16), c1);
    store_C<1>(T0, 32, 0,  c0, bm1_0);
    store_C<1>(T0, 32, 16, c1, bm1_1);
    __syncthreads();
    v16h A2 = frag_A<32>(T0, 32, 0);
    v8f d0 = {}, d1 = {};
    d0 = wmma16(A2, frag_B<32>(W + AO_WM2, 32, 0, 0), d0);
    d1 = wmma16(A2, frag_B<32>(W + AO_WM2, 32, 0, 16), d1);
    store_C<1>(T1, 32, 0,  d0, bm2_0);
    store_C<1>(T1, 32, 16, d1, bm2_1);
    __syncthreads();
    v16h A3 = frag_A<32>(T1, 32, 0);
    v8f e0 = {};
    e0 = wmma16(A3, frag_B<32>(W + AO_WMO, 16, 0, 0), e0);
    // write outs already in ordered layout (b, n, d); agent a -> position n
    const int n = (agt == AIDX) ? 0 : ((agt < AIDX) ? agt + 1 : agt);
#pragma unroll
    for (int v = 0; v < 8; ++v) {
      int r = v + lhalf * 8;
      wsO[((size_t)(b0 + r) * 16 + n) * 16 + lrow] = e0[v] + bmo;
    }
  }
}

// ---------------------------------------------------------------------------
// Critic params
// ---------------------------------------------------------------------------
struct CriticParams {
  const float *Wt, *bt, *Wp, *bp, *Wg, *bg;
  const float *Wm1, *bm1, *Wm2, *bm2, *Wm3, *bm3;
  const float *ln1g, *ln1b, *ln2g, *ln2b;
};

#define CO_WT   0
#define CO_BT   256
#define CO_WP   272
#define CO_BP   528
#define CO_WG   544
#define CO_BG   800
#define CO_WM1  816
#define CO_BM1  1840
#define CO_WM2  1872
#define CO_BM2  2896
#define CO_WM3  2928
#define CO_BM3  2960
#define CO_LN1G 2961
#define CO_LN1B 2977
#define CO_LN2G 2993
#define CO_LN2B 3009
#define CO_WTOT 3040
#define CO_PWAVE 3360

__global__ __launch_bounds__(128) void critic_kernel(const float* __restrict__ wsO,
                                                     CriticParams P,
                                                     float* __restrict__ out) {
  float* W = smem;
  const int tid = threadIdx.x;
  {
    const float* srcs[16] = {P.Wt,P.bt,P.Wp,P.bp,P.Wg,P.bg,P.Wm1,P.bm1,
                             P.Wm2,P.bm2,P.Wm3,P.bm3,P.ln1g,P.ln1b,P.ln2g,P.ln2b};
    const int offs[16] = {CO_WT,CO_BT,CO_WP,CO_BP,CO_WG,CO_BG,CO_WM1,CO_BM1,
                          CO_WM2,CO_BM2,CO_WM3,CO_BM3,CO_LN1G,CO_LN1B,CO_LN2G,CO_LN2B};
    const int ns[16]   = {256,16,256,16,256,16,1024,32,1024,32,32,1,16,16,16,16};
    for (int s = 0; s < 16; ++s)
      for (int i = tid; i < ns[s]; i += blockDim.x) W[offs[s] + i] = srcs[s][i];
  }
  __syncthreads();

  const int wave = tid >> 5;
  float* PWB = smem + CO_WTOT + wave * CO_PWAVE;
  float* OB   = PWB;          // [16 n][16 d] ordered actor outputs for one item
  float* TH   = PWB + 256;    // theta_pre [n][d]
  float* PH   = PWB + 512;    // phi_pre   [n][d]
  float* GG   = PWB + 768;    // g_pre     [n][d]
  float* SC2  = PWB + 1024;   // scores
  float* AT   = PWB + 1280;   // attn [d][e]
  float* AX   = PWB + 1536;   // allx [d][n]
  float* OTHv = PWB + 1792;   // [16]
  float* MRG  = PWB + 1808;   // [16][32]
  float* H1   = PWB + 2320;   // [16][32]
  float* H2   = PWB + 2832;   // [16][32]

  const int base = (blockIdx.x * 4 + wave) * 16;  // 16 batch items per wave
  const int l = lane_id(), lrow = l & 15, lhalf = l >> 4;

  // hoist loop-invariant projection weight fragments + biases
  const v16h Bt = frag_B<16>(W + CO_WT, 16, 0, 0);
  const v16h Bp = frag_B<16>(W + CO_WP, 16, 0, 0);
  const v16h Bg = frag_B<16>(W + CO_WG, 16, 0, 0);
  const float bbt = W[CO_BT + lrow], bbp = W[CO_BP + lrow], bbg = W[CO_BG + lrow];

  for (int it = 0; it < 16; ++it) {
    const int bi = base + it;
    for (int t = l; t < 256; t += 32) OB[t] = wsO[(size_t)bi * 256 + t];
    __syncthreads();

    // theta/phi/g = O @ W + b  (16x16 each)
    v16h AO = frag_A<16>(OB, 16, 0);
    v8f ct = {}, cp = {}, cg = {};
    ct = wmma16(AO, Bt, ct);
    cp = wmma16(AO, Bp, cp);
    cg = wmma16(AO, Bg, cg);
    store_C<0>(TH, 16, 0, ct, bbt);
    store_C<0>(PH, 16, 0, cp, bbp);
    store_C<0>(GG, 16, 0, cg, bbg);
    __syncthreads();

    // scores[d][e] = sum_n theta[n][d] * phi[n][e]  (A = TH^T, B = PH)
    v16h Ath = frag_AT<16>(TH, 16, 0);
    v8f sc = {};
    sc = wmma16(Ath, frag_B<16>(PH, 16, 0, 0), sc);
#pragma unroll
    for (int v = 0; v < 8; ++v) sc[v] *= INV_SCALE;
    store_C<0>(SC2, 16, 0, sc, 0.0f);
    __syncthreads();

    if (lhalf == 0) {  // softmax over e per row d
      float mx = -1e30f;
      for (int e = 0; e < 16; ++e) mx = fmaxf(mx, SC2[lrow * 16 + e]);
      float s = 0.0f;
      for (int e = 0; e < 16; ++e) { float ex = __expf(SC2[lrow * 16 + e] - mx); AT[lrow * 16 + e] = ex; s += ex; }
      float inv = 1.0f / s;
      for (int e = 0; e < 16; ++e) AT[lrow * 16 + e] *= inv;
    }
    __syncthreads();

    // allx[d][n] = sum_e attn[d][e] * g[n][e]  (B = GG^T)
    v16h Aat = frag_A<16>(AT, 16, 0);
    v8f ax = {};
    ax = wmma16(Aat, frag_BT<16>(GG, 16, 0), ax);
    store_C<0>(AX, 16, 0, ax, 0.0f);
    __syncthreads();

    if (lhalf == 0) {  // LN over d (per column n), gain/bias indexed by d; relu
      const int n = lrow;
      float m = 0.0f, v = 0.0f;
      for (int d = 0; d < 16; ++d) m += AX[d * 16 + n];
      m *= (1.0f / 16.0f);
      for (int d = 0; d < 16; ++d) { float t = AX[d * 16 + n] - m; v += t * t; }
      v *= (1.0f / 16.0f);
      float inv = rsqrtf(v + LN_EPS);
      for (int d = 0; d < 16; ++d) {
        float t = (AX[d * 16 + n] - m) * inv * W[CO_LN1G + d] + W[CO_LN1B + d];
        AX[d * 16 + n] = fmaxf(t, 0.0f);
      }
    }
    __syncthreads();

    if (l < 15) {  // a2 scores over the 15 "good" columns
      float acc = 0.0f;
      for (int d = 0; d < 16; ++d) acc += AX[d * 16 + 0] * AX[d * 16 + (l + 1)];
      SC2[l] = acc * INV_SCALE;
    }
    __syncthreads();
    {
      float mx = -1e30f;
      for (int j = 0; j < 15; ++j) mx = fmaxf(mx, SC2[j]);
      float s = 0.0f;
      float a2[15];
      for (int j = 0; j < 15; ++j) { a2[j] = __expf(SC2[j] - mx); s += a2[j]; }
      float inv = 1.0f / s;
      if (lhalf == 0) {
        const int d = lrow;
        float acc = 0.0f;
        for (int j = 0; j < 15; ++j) acc += a2[j] * inv * AX[d * 16 + (j + 1)];
        OTHv[d] = acc;
      }
    }
    __syncthreads();
    if (lhalf == 0) {  // LN over oth, build merged row
      float m = 0.0f, v = 0.0f;
      for (int d = 0; d < 16; ++d) m += OTHv[d];
      m *= (1.0f / 16.0f);
      for (int d = 0; d < 16; ++d) { float t = OTHv[d] - m; v += t * t; }
      v *= (1.0f / 16.0f);
      float inv = rsqrtf(v + LN_EPS);
      const int d = lrow;
      MRG[it * 32 + d] = OB[d];  // ordered[0] row (n = 0)
      MRG[it * 32 + 16 + d] = fmaxf((OTHv[d] - m) * inv * W[CO_LN2G + d] + W[CO_LN2B + d], 0.0f);
    }
    __syncthreads();
  }

  // merged MLP over 16 items: 32 -> 32 -> 32 -> 1, leaky_relu(0.01)
  {
    const float bm1_0 = W[CO_BM1 + lrow], bm1_1 = W[CO_BM1 + 16 + lrow];
    const float bm2_0 = W[CO_BM2 + lrow], bm2_1 = W[CO_BM2 + 16 + lrow];
    v16h Am = frag_A<32>(MRG, 32, 0);
    v8f h0 = {}, h1 = {};
    h0 = wmma16(Am, frag_B<32>(W + CO_WM1, 32, 0, 0), h0);
    h1 = wmma16(Am, frag_B<32>(W + CO_WM1, 32, 0, 16), h1);
    store_C<2>(H1, 32, 0,  h0, bm1_0);
    store_C<2>(H1, 32, 16, h1, bm1_1);
    __syncthreads();
    v16h Ah = frag_A<32>(H1, 32, 0);
    v8f g0 = {}, g1 = {};
    g0 = wmma16(Ah, frag_B<32>(W + CO_WM2, 32, 0, 0), g0);
    g1 = wmma16(Ah, frag_B<32>(W + CO_WM2, 32, 0, 16), g1);
    store_C<2>(H2, 32, 0,  g0, bm2_0);
    store_C<2>(H2, 32, 16, g1, bm2_1);
    __syncthreads();
    if (lhalf == 0) {
      float acc = W[CO_BM3];
      for (int k = 0; k < 32; ++k) acc += H2[lrow * 32 + k] * W[CO_WM3 + k];
      out[base + lrow] = acc;
    }
  }
}

// ---------------------------------------------------------------------------
extern "C" void kernel_launch(void* const* d_in, const int* in_sizes, int n_in,
                              void* d_out, int out_size, void* d_ws, size_t ws_size,
                              hipStream_t stream) {
  (void)in_sizes; (void)n_in; (void)out_size; (void)ws_size;
  const float* obs = (const float*)d_in[0];
  const float* act = (const float*)d_in[1];
  // params dict insertion order: (name_w, name_b) for each linear, then LN g/b
  int p = 2;
  ActorParams AP;
  AP.Wself1 = (const float*)d_in[p++]; AP.bself1 = (const float*)d_in[p++];
  AP.Wself2 = (const float*)d_in[p++]; AP.bself2 = (const float*)d_in[p++];
  AP.Wfood1 = (const float*)d_in[p++]; AP.bfood1 = (const float*)d_in[p++];
  AP.Wfood2 = (const float*)d_in[p++]; AP.bfood2 = (const float*)d_in[p++];
  AP.Woth1  = (const float*)d_in[p++]; AP.both1  = (const float*)d_in[p++];
  AP.Woth2  = (const float*)d_in[p++]; AP.both2  = (const float*)d_in[p++];
  AP.Wm1    = (const float*)d_in[p++]; AP.bm1    = (const float*)d_in[p++];
  AP.Wm2    = (const float*)d_in[p++]; AP.bm2    = (const float*)d_in[p++];
  AP.Wmo    = (const float*)d_in[p++]; AP.bmo    = (const float*)d_in[p++];
  CriticParams CP;
  CP.Wt  = (const float*)d_in[p++]; CP.bt  = (const float*)d_in[p++];
  CP.Wp  = (const float*)d_in[p++]; CP.bp  = (const float*)d_in[p++];
  CP.Wg  = (const float*)d_in[p++]; CP.bg  = (const float*)d_in[p++];
  CP.Wm1 = (const float*)d_in[p++]; CP.bm1 = (const float*)d_in[p++];
  CP.Wm2 = (const float*)d_in[p++]; CP.bm2 = (const float*)d_in[p++];
  CP.Wm3 = (const float*)d_in[p++]; CP.bm3 = (const float*)d_in[p++];
  AP.flng = (const float*)d_in[p++]; AP.flnb = (const float*)d_in[p++];
  AP.olng = (const float*)d_in[p++]; AP.olnb = (const float*)d_in[p++];
  CP.ln1g = (const float*)d_in[p++]; CP.ln1b = (const float*)d_in[p++];
  CP.ln2g = (const float*)d_in[p++]; CP.ln2b = (const float*)d_in[p++];

  float* wsO = (float*)d_ws;  // (BATCH, 16 ordered agents, 16) f32 = 16 MB

  const size_t ashm = (size_t)(AO_WTOT + 4 * AO_PWAVE) * sizeof(float);   // ~113 KB
  const size_t cshm = (size_t)(CO_WTOT + 4 * CO_PWAVE) * sizeof(float);   // ~64 KB

  // actor: 16*BATCH rows / 16 rows-per-wave / 4 waves-per-block = 4096 blocks
  actor_kernel<<<dim3((NAG * BATCH_N) / (16 * 4)), dim3(128), ashm, stream>>>(obs, act, AP, wsO);
  // critic: BATCH items / 16 items-per-wave / 4 waves-per-block = 256 blocks
  critic_kernel<<<dim3(BATCH_N / (16 * 4)), dim3(128), cshm, stream>>>(wsO, CP, (float*)d_out);
}
